// MultiHead_Group_Query_Attention_22033182228930
// MI455X (gfx1250) — matrix operand order repaired and checked by reading, not compiled, not run on hardware
//
#include <hip/hip_runtime.h>
#include <hip/hip_bf16.h>

// ---------------------------------------------------------------------------
// CDNA5 (gfx1250) wave32 WMMA implementation of causal GQA attention.
//   x[4096,1024]f32, w_q[1024,1024], w_k/w_v[1024,256], w_o[1024,1024]
//   out[4096,1024]f32
// All matmuls via v_wmma_f32_16x16x32_bf16 (bf16 in, f32 accumulate).
// Attention computes S^T = K Q^T so that P = exp(S^T) lands directly in the
// WMMA A-fragment layout for P*V (in-lane bf16 packs, no LDS re-stripe).
// Softmax runs in the log2 domain (scale*log2e folded into Q at projection),
// and causal masking is confined to a single tail block outside the hot loop.
// ---------------------------------------------------------------------------

typedef __attribute__((ext_vector_type(16))) __bf16 v16bf;
typedef __attribute__((ext_vector_type(8)))  float  v8f;

union FragB16 { v16bf v; uint4 q[2]; unsigned int u[8]; };

__device__ __forceinline__ unsigned short f32_to_bf16_raw(float f) {
  unsigned int u = __float_as_uint(f);
  u += 0x7FFFu + ((u >> 16) & 1u);       // round-to-nearest-even
  return (unsigned short)(u >> 16);
}

// truncating pack for attention probabilities (cheap: and+shift+or / v_perm)
__device__ __forceinline__ unsigned int pack_bf16_trunc(float lo, float hi) {
  return (__float_as_uint(hi) & 0xffff0000u) | (__float_as_uint(lo) >> 16);
}

// ---------------------------------------------------------------------------
// Elementwise conversion kernels
// ---------------------------------------------------------------------------
__global__ void k_f32_to_bf16(const float* __restrict__ src,
                              unsigned short* __restrict__ dst, int n) {
  int i = blockIdx.x * blockDim.x + threadIdx.x;
  if (i < n) dst[i] = f32_to_bf16_raw(src[i]);
}

// src[rows][cols] f32 -> dst[cols][rows] bf16  (transpose)
__global__ void k_transpose_f32_to_bf16(const float* __restrict__ src,
                                        unsigned short* __restrict__ dst,
                                        int rows, int cols) {
  int i = blockIdx.x * blockDim.x + threadIdx.x;
  if (i < rows * cols) {
    int r = i / cols, c = i % cols;
    dst[(size_t)c * rows + r] = f32_to_bf16_raw(src[i]);
  }
}

// ---------------------------------------------------------------------------
// Wave32 WMMA GEMM:  C[M,N] = oscale * (A[M,K] * Bt[N,K]^T)
// One wave computes a 32x64 tile (2 M-subtiles x 4 N-subtiles).
// MODE: 0 = f32 row-major, 1 = bf16 row-major, 2 = bf16 transposed (out[n][m])
// grid = (N/64, M/32), block = 32
// ---------------------------------------------------------------------------
template <int MODE>
__device__ __forceinline__ void store_tile(void* out, v8f c, int m0, int n,
                                           int half, int ldo, float oscale) {
#pragma unroll
  for (int v = 0; v < 8; ++v) {
    int m = m0 + v + 8 * half;          // C/D layout: VGPR v -> row v + 8*(lane/16)
    float val = c[v] * oscale;
    if (MODE == 0)
      ((float*)out)[(size_t)m * ldo + n] = val;
    else if (MODE == 1)
      ((unsigned short*)out)[(size_t)m * ldo + n] = f32_to_bf16_raw(val);
    else
      ((unsigned short*)out)[(size_t)n * ldo + m] = f32_to_bf16_raw(val);
  }
}

template <int MODE>
__global__ void __launch_bounds__(32)
k_gemm_bf16(const unsigned short* __restrict__ A,
            const unsigned short* __restrict__ Bt,
            void* __restrict__ out,
            int K, int lda, int ldb, int ldo, float oscale) {
  const int lane = threadIdx.x;
  const int ln   = lane & 15;
  const int half = lane >> 4;
  const int m0   = blockIdx.y * 32;
  const int n0   = blockIdx.x * 64;
  const int aoff = half * 8;    // A frag: K = {aoff..aoff+7} U {aoff+16..aoff+23}
  const int boff = half * 8;    // B frag: K = {boff..boff+7} U {boff+16..boff+23}

  const unsigned short* arow0 = A + (size_t)(m0 + ln) * lda;
  const unsigned short* arow1 = A + (size_t)(m0 + 16 + ln) * lda;

  v8f acc00 = {}, acc01 = {}, acc02 = {}, acc03 = {};
  v8f acc10 = {}, acc11 = {}, acc12 = {}, acc13 = {};
  for (int k0 = 0; k0 < K; k0 += 32) {
    FragB16 a0, a1;
    a0.q[0] = *(const uint4*)(arow0 + k0 + aoff);
    a0.q[1] = *(const uint4*)(arow0 + k0 + aoff + 16);
    a1.q[0] = *(const uint4*)(arow1 + k0 + aoff);
    a1.q[1] = *(const uint4*)(arow1 + k0 + aoff + 16);

#define GEMM_STEP(T, ACC0, ACC1)                                              \
    {                                                                         \
      const unsigned short* brow =                                            \
          Bt + (size_t)(n0 + (T)*16 + ln) * ldb + k0 + boff;                  \
      FragB16 b;                                                              \
      b.q[0] = *(const uint4*)(brow);                                         \
      b.q[1] = *(const uint4*)(brow + 16);                                    \
      ACC0 = __builtin_amdgcn_wmma_f32_16x16x32_bf16(                         \
          false, a0.v, false, b.v, (short)0, ACC0, false, false);             \
      ACC1 = __builtin_amdgcn_wmma_f32_16x16x32_bf16(                         \
          false, a1.v, false, b.v, (short)0, ACC1, false, false);             \
    }
    GEMM_STEP(0, acc00, acc10)
    GEMM_STEP(1, acc01, acc11)
    GEMM_STEP(2, acc02, acc12)
    GEMM_STEP(3, acc03, acc13)
#undef GEMM_STEP
  }

  store_tile<MODE>(out, acc00, m0,      n0 +  0 + ln, half, ldo, oscale);
  store_tile<MODE>(out, acc01, m0,      n0 + 16 + ln, half, ldo, oscale);
  store_tile<MODE>(out, acc02, m0,      n0 + 32 + ln, half, ldo, oscale);
  store_tile<MODE>(out, acc03, m0,      n0 + 48 + ln, half, ldo, oscale);
  store_tile<MODE>(out, acc10, m0 + 16, n0 +  0 + ln, half, ldo, oscale);
  store_tile<MODE>(out, acc11, m0 + 16, n0 + 16 + ln, half, ldo, oscale);
  store_tile<MODE>(out, acc12, m0 + 16, n0 + 32 + ln, half, ldo, oscale);
  store_tile<MODE>(out, acc13, m0 + 16, n0 + 48 + ln, half, ldo, oscale);
}

// ---------------------------------------------------------------------------
// Flash-style causal GQA attention, transposed-score formulation.
// One wave per (16-row q block, head).  grid = (256, 16), block = 32.
//   Qb [4096][1024] bf16 (pre-scaled by log2e/sqrt(hd)), Kb [4096][256] bf16,
//   Vt [256][4096] bf16 (V^T), Ctx[4096][1024] bf16 ([s][h*64+d])
// Hot loop: full 64-key blocks, no masking. Tail: exactly one masked block.
// ---------------------------------------------------------------------------
__global__ void __launch_bounds__(32)
k_attn(const unsigned short* __restrict__ Qb,
       const unsigned short* __restrict__ Kb,
       const unsigned short* __restrict__ Vt,
       unsigned short* __restrict__ Ctx) {
  const int lane = threadIdx.x;
  const int ln   = lane & 15;
  const int half = lane >> 4;
  const int qb   = blockIdx.x;   // q block (16 rows)
  const int h    = blockIdx.y;   // query head
  const int kv   = h >> 2;       // GQA group of 4
  const int aoff = half * 8;
  const int boff = half * 8;

  // Q enters as the B operand of S' = K Q^T: lane holds query ln,
  // head-dim {boff..boff+7} U {boff+16..boff+23}. Two k-steps (hd=64).
  const unsigned short* qrow = Qb + (size_t)(qb * 16 + ln) * 1024 + h * 64;
  FragB16 qf0, qf1;
  qf0.q[0] = *(const uint4*)(qrow + boff);
  qf0.q[1] = *(const uint4*)(qrow + boff + 16);
  qf1.q[0] = *(const uint4*)(qrow + 32 + boff);
  qf1.q[1] = *(const uint4*)(qrow + 32 + boff + 16);

  const unsigned short* kbase_p = Kb + kv * 64;             // + row*256
  const unsigned short* vbase_p = Vt + (size_t)(kv * 64 + ln) * 4096;

  v8f o0 = {}, o1 = {}, o2 = {}, o3 = {};   // O: (q x 64d), q = vgpr, d = lane
  float rowm = -1e30f;                       // log2-domain stats for query ln
  float rowl = 0.0f;

  const int qmin  = qb * 16;
  const int qmax  = qb * 16 + 15;
  const int qlane = qb * 16 + ln;            // this lane's query row
  const int nfull = qmin >> 6;               // full (unmasked) 64-key blocks

// --- compute one 16-key score tile (no mask), result into S ---
#define SCORE_TILE_NOMASK(KT, S)                                              \
    {                                                                         \
      const unsigned short* kr = kbase_p + (size_t)((KT) + ln) * 256;         \
      FragB16 kf0, kf1;                                                       \
      kf0.q[0] = *(const uint4*)(kr + aoff);                                  \
      kf0.q[1] = *(const uint4*)(kr + aoff + 16);                             \
      kf1.q[0] = *(const uint4*)(kr + 32 + aoff);                             \
      kf1.q[1] = *(const uint4*)(kr + 32 + aoff + 16);                        \
      v8f t = {};                                                             \
      t = __builtin_amdgcn_wmma_f32_16x16x32_bf16(                            \
          false, kf0.v, false, qf0.v, (short)0, t, false, false);             \
      t = __builtin_amdgcn_wmma_f32_16x16x32_bf16(                            \
          false, kf1.v, false, qf1.v, (short)0, t, false, false);             \
      S = t;                                                                  \
    }

// --- softmax + P-pack + PV accumulate for one 64-key block at KBASE ---
#define SOFTMAX_PV(KBASE)                                                     \
    {                                                                         \
      float bm;                                                               \
      {                                                                       \
        v8f m01, m23;                                                         \
        _Pragma("unroll")                                                     \
        for (int v = 0; v < 8; ++v) m01[v] = fmaxf(s0[v], s1[v]);             \
        _Pragma("unroll")                                                     \
        for (int v = 0; v < 8; ++v) m23[v] = fmaxf(s2[v], s3[v]);             \
        bm = fmaxf(m01[0], m23[0]);                                           \
        _Pragma("unroll")                                                     \
        for (int v = 1; v < 8; ++v) bm = fmaxf(bm, fmaxf(m01[v], m23[v]));    \
      }                                                                       \
      bm = fmaxf(bm, __shfl_xor(bm, 16, 32));                                 \
      const float mnew = fmaxf(rowm, bm);                                     \
      const float corr = __builtin_amdgcn_exp2f(rowm - mnew);                 \
      rowm = mnew;                                                            \
      float rs = 0.0f;                                                        \
      _Pragma("unroll")                                                       \
      for (int v = 0; v < 8; ++v) {                                           \
        s0[v] = __builtin_amdgcn_exp2f(s0[v] - mnew);                         \
        s1[v] = __builtin_amdgcn_exp2f(s1[v] - mnew);                         \
        s2[v] = __builtin_amdgcn_exp2f(s2[v] - mnew);                         \
        s3[v] = __builtin_amdgcn_exp2f(s3[v] - mnew);                         \
        rs += (s0[v] + s1[v]) + (s2[v] + s3[v]);                              \
      }                                                                       \
      rs += __shfl_xor(rs, 16, 32);                                           \
      rowl = rowl * corr + rs;                                                \
      _Pragma("unroll")                                                       \
      for (int v = 0; v < 8; ++v) {                                           \
        float cb = __shfl(corr, v + 8 * half, 32);                            \
        o0[v] *= cb; o1[v] *= cb; o2[v] *= cb; o3[v] *= cb;                   \
      }                                                                       \
      FragB16 pf0, pf1;                                                       \
      _Pragma("unroll")                                                       \
      for (int j = 0; j < 4; ++j) {                                           \
        pf0.u[j]     = pack_bf16_trunc(s0[2 * j], s0[2 * j + 1]);             \
        pf0.u[4 + j] = pack_bf16_trunc(s1[2 * j], s1[2 * j + 1]);             \
        pf1.u[j]     = pack_bf16_trunc(s2[2 * j], s2[2 * j + 1]);             \
        pf1.u[4 + j] = pack_bf16_trunc(s3[2 * j], s3[2 * j + 1]);             \
      }                                                                       \
      _Pragma("unroll")                                                       \
      for (int dt = 0; dt < 4; ++dt) {                                        \
        const unsigned short* vr = vbase_p + (size_t)(dt * 16) * 4096 + (KBASE); \
        FragB16 vf0, vf1;                                                     \
        vf0.q[0] = *(const uint4*)(vr + boff);                                \
        vf0.q[1] = *(const uint4*)(vr + boff + 16);                           \
        vf1.q[0] = *(const uint4*)(vr + 32 + boff);                           \
        vf1.q[1] = *(const uint4*)(vr + 32 + boff + 16);                      \
        v8f oo = dt == 0 ? o0 : dt == 1 ? o1 : dt == 2 ? o2 : o3;             \
        oo = __builtin_amdgcn_wmma_f32_16x16x32_bf16(                         \
            false, pf0.v, false, vf0.v, (short)0, oo, false, false);          \
        oo = __builtin_amdgcn_wmma_f32_16x16x32_bf16(                         \
            false, pf1.v, false, vf1.v, (short)0, oo, false, false);          \
        if (dt == 0) o0 = oo; else if (dt == 1) o1 = oo;                      \
        else if (dt == 2) o2 = oo; else o3 = oo;                              \
      }                                                                       \
    }

  // ---- hot loop: full 64-key blocks, zero mask logic ----
  for (int kbk = 0; kbk < nfull; ++kbk) {
    const int kbase = kbk * 64;
    v8f s0, s1, s2, s3;
    SCORE_TILE_NOMASK(kbase +  0, s0)
    SCORE_TILE_NOMASK(kbase + 16, s1)
    SCORE_TILE_NOMASK(kbase + 32, s2)
    SCORE_TILE_NOMASK(kbase + 48, s3)
    SOFTMAX_PV(kbase)
  }

  // ---- tail: exactly one partially masked 64-key block ----
  {
    const int kbase = nfull * 64;
    v8f s0, s1, s2, s3;
#define SCORE_TILE_MASK(T, S)                                                 \
    {                                                                         \
      const int kt = kbase + (T) * 16;                                        \
      if (kt <= qmax) { /* wave-uniform: EXEC stays full for WMMA */          \
        SCORE_TILE_NOMASK(kt, S)                                              \
        if (kt + 15 > qmin) {                                                 \
          _Pragma("unroll")                                                   \
          for (int v = 0; v < 8; ++v) {                                       \
            int key = kt + v + 8 * half;                                      \
            if (key > qlane) S[v] = -1e30f;                                   \
          }                                                                   \
        }                                                                     \
      } else {                                                                \
        _Pragma("unroll")                                                     \
        for (int v = 0; v < 8; ++v) S[v] = -1e30f;                            \
      }                                                                       \
    }
    SCORE_TILE_MASK(0, s0)
    SCORE_TILE_MASK(1, s1)
    SCORE_TILE_MASK(2, s2)
    SCORE_TILE_MASK(3, s3)
#undef SCORE_TILE_MASK
    SOFTMAX_PV(kbase)
  }
#undef SOFTMAX_PV
#undef SCORE_TILE_NOMASK

  // ---- epilogue: normalize (broadcast 1/l per O row) and store bf16 ctx ----
  const float linv = 1.0f / rowl;
#pragma unroll
  for (int v = 0; v < 8; ++v) {
    float lb = __shfl(linv, v + 8 * half, 32);
    int m = qb * 16 + v + 8 * half;
    unsigned short* crow = Ctx + (size_t)m * 1024 + h * 64;
    crow[ln]      = f32_to_bf16_raw(o0[v] * lb);
    crow[16 + ln] = f32_to_bf16_raw(o1[v] * lb);
    crow[32 + ln] = f32_to_bf16_raw(o2[v] * lb);
    crow[48 + ln] = f32_to_bf16_raw(o3[v] * lb);
  }
}

// ---------------------------------------------------------------------------
// Host launcher
// ---------------------------------------------------------------------------
extern "C" void kernel_launch(void* const* d_in, const int* in_sizes, int n_in,
                              void* d_out, int out_size, void* d_ws,
                              size_t ws_size, hipStream_t stream) {
  const float* x  = (const float*)d_in[0];
  const float* wq = (const float*)d_in[1];
  const float* wk = (const float*)d_in[2];
  const float* wv = (const float*)d_in[3];
  const float* wo = (const float*)d_in[4];
  float* out = (float*)d_out;

  char* ws = (char*)d_ws;
  const size_t MB = 1048576;
  unsigned short* xb  = (unsigned short*)(ws + 0 * MB);             // 8 MB
  unsigned short* qbv = (unsigned short*)(ws + 8 * MB);             // 8 MB
  unsigned short* kbv = (unsigned short*)(ws + 16 * MB);            // 2 MB
  unsigned short* vtb = (unsigned short*)(ws + 18 * MB);            // 2 MB  V^T
  unsigned short* ctx = (unsigned short*)(ws + 20 * MB);            // 8 MB
  unsigned short* wqt = (unsigned short*)(ws + 28 * MB);            // 2 MB
  unsigned short* wkt = (unsigned short*)(ws + 30 * MB);            // .5 MB
  unsigned short* wvt = (unsigned short*)(ws + (30 * MB + MB / 2)); // .5 MB
  unsigned short* wot = (unsigned short*)(ws + 31 * MB);            // 2 MB

  // Conversions
  k_f32_to_bf16<<<4194304 / 256, 256, 0, stream>>>(x, xb, 4194304);
  k_transpose_f32_to_bf16<<<1048576 / 256, 256, 0, stream>>>(wq, wqt, 1024, 1024);
  k_transpose_f32_to_bf16<<<262144 / 256, 256, 0, stream>>>(wk, wkt, 1024, 256);
  k_transpose_f32_to_bf16<<<262144 / 256, 256, 0, stream>>>(wv, wvt, 1024, 256);
  k_transpose_f32_to_bf16<<<1048576 / 256, 256, 0, stream>>>(wo, wot, 1024, 1024);

  // Projections (32x64 tile per wave). Q carries log2(e)/sqrt(head_dim) so
  // the attention softmax can run natively in exp2.
  const float qscale = 0.125f * 1.44269504088896340736f;
  dim3 gq(1024 / 64, 4096 / 32);  // 16 x 128
  dim3 gk(256 / 64, 4096 / 32);   // 4  x 128
  k_gemm_bf16<1><<<gq, 32, 0, stream>>>(xb, wqt, qbv, 1024, 1024, 1024, 1024, qscale);
  k_gemm_bf16<1><<<gk, 32, 0, stream>>>(xb, wkt, kbv, 1024, 1024, 1024, 256, 1.0f);
  k_gemm_bf16<2><<<gk, 32, 0, stream>>>(xb, wvt, vtb, 1024, 1024, 1024, 4096, 1.0f);

  // Attention
  dim3 ga(256, 16);
  k_attn<<<ga, 32, 0, stream>>>(qbv, kbv, vtb, ctx);

  // Output projection (f32 result)
  k_gemm_bf16<0><<<gq, 32, 0, stream>>>(ctx, wot, out, 1024, 1024, 1024, 1024, 1.0f);
}
